// ExpandFormerV15Complete_77438260347447
// MI455X (gfx1250) — matrix operand order repaired and checked by reading, not compiled, not run on hardware
//
#include <hip/hip_runtime.h>
#include <hip/hip_bf16.h>
#include <math.h>

typedef __attribute__((ext_vector_type(16))) __bf16 v16bf;
typedef __attribute__((ext_vector_type(8)))  float  v8f;

#define BASE   64
#define HID    128
#define NDOM   8
#define CORR_SCALE 0.1f

// ---- workspace layout (bytes) ----
// [0,32)                : counts[NDOM] (int)
// [256, 256+4*NDOM*ntok): lists (int), stride ntok per domain
// then bf16 swizzled W1 (NDOM*BASE*HID) and W2 (NDOM*HID*BASE)

static __device__ __forceinline__ v8f wmma_bf16(v16bf a, v16bf b, v8f c) {
    return __builtin_amdgcn_wmma_f32_16x16x32_bf16(
        /*neg_a=*/false, a, /*neg_b=*/false, b,
        /*c_mod=*/(short)0, c, /*reuse_a=*/false, /*reuse_b=*/false);
}

// Branch-free GELU (tanh form rewritten as sigmoid), all-hardware ops:
//   gelu(x) ~= x * sigmoid(1.5957691(x + 0.044715 x^3))
//           =  x * rcp(1 + exp2(-2.3022078(x + 0.044715 x^3)))
// One v_exp_f32 + one v_rcp_f32; for this workload |x| <~ 0.01 this agrees
// with the exact erf GELU to O(x^5) ~ 1e-10, and the correction it feeds is
// ~1e-5 of the output magnitude.
static __device__ __forceinline__ float gelu_fast(float v) {
    float p = __builtin_fmaf(0.044715f * v, v * v, v);
    float e = __builtin_amdgcn_exp2f(-2.302207767f * p);   // 1.5957691 * log2(e)
    return v * __builtin_amdgcn_rcpf(1.0f + e);
}

__global__ void zero_counts_kernel(int* counts) {
    if (threadIdx.x < NDOM) counts[threadIdx.x] = 0;
}

// one thread per token: find its (unique) domain, append to that domain's list
__global__ void classify_kernel(const int* __restrict__ x,
                                const float* __restrict__ member,
                                int* __restrict__ counts,
                                int* __restrict__ lists, int ntok) {
    int t = blockIdx.x * blockDim.x + threadIdx.x;
    if (t >= ntok) return;
    int tok = x[t];
    const float4* mrow = (const float4*)(member + (size_t)tok * NDOM);
    float4 m0 = mrow[0], m1 = mrow[1];
    int dom = -1;
    if (m0.x > 0.5f) dom = 0;
    if (m0.y > 0.5f) dom = 1;
    if (m0.z > 0.5f) dom = 2;
    if (m0.w > 0.5f) dom = 3;
    if (m1.x > 0.5f) dom = 4;
    if (m1.y > 0.5f) dom = 5;
    if (m1.z > 0.5f) dom = 6;
    if (m1.w > 0.5f) dom = 7;
    if (dom >= 0) {
        int pos = atomicAdd(&counts[dom], 1);
        lists[dom * ntok + pos] = t;
    }
}

// out[t,:] = embed[x[t],:]   (one float4 per thread, 16 threads/token)
__global__ void copy_base_kernel(const int* __restrict__ x,
                                 const float* __restrict__ embed,
                                 float* __restrict__ out, int ntok) {
    int i = blockIdx.x * blockDim.x + threadIdx.x;
    if (i >= ntok * (BASE / 4)) return;
    int t = i >> 4;
    int j = i & 15;
    const float4* src = (const float4*)(embed + (size_t)x[t] * BASE);
    ((float4*)(out + (size_t)t * BASE))[j] = src[j];
}

// Convert W1/W2 (f32) into bf16 WMMA B-fragment layout:
// fragment = 32 lanes x 16 bf16; lane l holds column N = nb*16 + (l&15),
// element e holds K = kb*32 + (l>>4)*8 + e (e<8), kb*32 + 16 + (l>>4)*8 + (e-8) (e>=8).
__global__ void prep_weights_kernel(const float* __restrict__ W1,
                                    const float* __restrict__ W2,
                                    __bf16* __restrict__ w1s,
                                    __bf16* __restrict__ w2s) {
    int e = blockIdx.x * blockDim.x + threadIdx.x;
    const int W1N = NDOM * 2 * 8 * 512;   // 65536 elements
    const int W2N = NDOM * 4 * 4 * 512;   // 65536 elements
    if (e < W1N) {
        int slot = e & 511, frag = e >> 9;
        int lane = slot >> 4, elem = slot & 15;
        int nb = frag & 7, kb = (frag >> 3) & 1, d = frag >> 4;
        int n = nb * 16 + (lane & 15);
        int k = kb * 32 + (lane >> 4) * 8 + (elem & 7) + (elem >> 3) * 16;   // k in [0,64)
        w1s[e] = (__bf16)W1[(d * BASE + k) * HID + n];
        return;
    }
    int e2 = e - W1N;
    if (e2 < W2N) {
        int slot = e2 & 511, frag = e2 >> 9;
        int lane = slot >> 4, elem = slot & 15;
        int nb = frag & 3, kb = (frag >> 2) & 3, d = frag >> 4;
        int n = nb * 16 + (lane & 15);                // n in [0,64)
        int k = kb * 32 + (lane >> 4) * 8 + (elem & 7) + (elem >> 3) * 16;    // k in [0,128)
        w2s[e2] = (__bf16)W2[(d * HID + k) * BASE + n];
    }
}

// One wave (32 threads) per block. blockIdx.y = domain, blockIdx.x strides tiles.
// Per 16-token tile: GEMM1 (16x64 @ 64x128) -> GELU -> LDS restage ->
// GEMM2 (16x128 @ 128x64) -> out += 0.1*corr.
__global__ __launch_bounds__(32)
void expert_kernel(float* __restrict__ out,
                   const int* __restrict__ counts,
                   const int* __restrict__ lists,
                   const __bf16* __restrict__ w1s,
                   const __bf16* __restrict__ w2s, int ntok) {
    __shared__ __bf16 hidS[16][HID];   // 4 KB, single-wave workgroup

    const int dom  = blockIdx.y;
    const int cnt  = counts[dom];
    const int lane = threadIdx.x;            // wave32
    const int mrow  = lane & 15;             // A-matrix: M = lane%16
    const int khalf = lane >> 4;             // A-matrix: K-half select
    const int ncol  = lane & 15;             // C/D: N = lane%16
    const int mbase = (lane >> 4) * 8;       // C/D: rows mbase..mbase+7
    const int* list = lists + dom * ntok;

    for (int tile = blockIdx.x; tile * 16 < cnt; tile += gridDim.x) {
        // ---- load A (16x64 of h, bf16) per ISA 16-bit A layout ----
        int idx = tile * 16 + mrow;
        int t = list[idx < cnt ? idx : tile * 16];
        const float* hrowp = out + (size_t)t * BASE;   // base already copied
        v16bf a0, a1;
#pragma unroll
        for (int kb = 0; kb < 2; ++kb) {
            const float* p0 = hrowp + kb * 32 + khalf * 8;
            const float* p1 = hrowp + kb * 32 + 16 + khalf * 8;
            v16bf a;
#pragma unroll
            for (int q = 0; q < 8; ++q) {
                a[q]     = (__bf16)p0[q];
                a[q + 8] = (__bf16)p1[q];
            }
            if (kb == 0) a0 = a; else a1 = a;
        }

        // ---- GEMM1 + GELU, stage hid into LDS ----
#pragma unroll
        for (int nb = 0; nb < 8; ++nb) {
            const v16bf b0 = *(const v16bf*)(w1s + (((dom * 2 + 0) * 8 + nb) * 512) + lane * 16);
            const v16bf b1 = *(const v16bf*)(w1s + (((dom * 2 + 1) * 8 + nb) * 512) + lane * 16);
            v8f c = {};
            c = wmma_bf16(a0, b0, c);
            c = wmma_bf16(a1, b1, c);
#pragma unroll
            for (int r = 0; r < 8; ++r) {
                hidS[mbase + r][nb * 16 + ncol] = (__bf16)gelu_fast(c[r]);
            }
        }
        __syncthreads();   // single-wave WG: orders LDS write->read across lanes

        // ---- restage hid as A fragments (16x128 bf16, 4 K-blocks) ----
        v16bf a2[4];
#pragma unroll
        for (int kb = 0; kb < 4; ++kb) {
            const __bf16* q0 = &hidS[mrow][kb * 32 + khalf * 8];
            const __bf16* q1 = &hidS[mrow][kb * 32 + 16 + khalf * 8];
#pragma unroll
            for (int q = 0; q < 8; ++q) {
                a2[kb][q]     = q0[q];
                a2[kb][q + 8] = q1[q];
            }
        }
        __syncthreads();

        // ---- GEMM2 + scatter-add correction ----
#pragma unroll
        for (int nb = 0; nb < 4; ++nb) {
            v8f c = {};
#pragma unroll
            for (int kb = 0; kb < 4; ++kb) {
                const v16bf b = *(const v16bf*)(w2s + (((dom * 4 + kb) * 4 + nb) * 512) + lane * 16);
                c = wmma_bf16(a2[kb], b, c);
            }
#pragma unroll
            for (int r = 0; r < 8; ++r) {
                int idx2 = tile * 16 + mbase + r;
                if (idx2 < cnt) {
                    int t2 = list[idx2];
                    float* o = out + (size_t)t2 * BASE + nb * 16 + ncol;
                    *o += CORR_SCALE * c[r];
                }
            }
        }
    }
}

extern "C" void kernel_launch(void* const* d_in, const int* in_sizes, int n_in,
                              void* d_out, int out_size, void* d_ws, size_t ws_size,
                              hipStream_t stream) {
    const int*   x      = (const int*)d_in[0];
    const float* embed  = (const float*)d_in[1];
    const float* W1     = (const float*)d_in[2];
    const float* W2     = (const float*)d_in[3];
    const float* member = (const float*)d_in[4];
    float* out = (float*)d_out;
    const int ntok = in_sizes[0];   // B*S = 32768

    char* ws = (char*)d_ws;
    int*    counts = (int*)(ws + 0);
    int*    lists  = (int*)(ws + 256);
    size_t  w1off  = 256 + (size_t)NDOM * ntok * sizeof(int);
    w1off = (w1off + 255) & ~(size_t)255;
    __bf16* w1s = (__bf16*)(ws + w1off);
    __bf16* w2s = (__bf16*)(ws + w1off + (size_t)NDOM * BASE * HID * sizeof(__bf16));

    zero_counts_kernel<<<1, 32, 0, stream>>>(counts);
    classify_kernel<<<(ntok + 255) / 256, 256, 0, stream>>>(x, member, counts, lists, ntok);
    copy_base_kernel<<<(ntok * (BASE / 4) + 255) / 256, 256, 0, stream>>>(x, embed, out, ntok);
    prep_weights_kernel<<<(2 * NDOM * BASE * HID + 255) / 256, 256, 0, stream>>>(W1, W2, w1s, w2s);
    expert_kernel<<<dim3(256, NDOM), 32, 0, stream>>>(out, counts, lists, w1s, w2s, ntok);
}
